// PointShufflerSetAbstraction_54159537602743
// MI455X (gfx1250) — matrix operand order repaired and checked by reading, not compiled, not run on hardware
//
#include <hip/hip_runtime.h>
#include <hip/hip_bf16.h>

// ---------------------------------------------------------------------------
// PointShufflerSetAbstraction for MI455X (gfx1250, wave32, WMMA).
//   pack feats(f16) + pack weights(WMMA A-order f16 tiles) ->
//   gemm0(WMMA 6->32) -> bnfin -> gemm(32->64) -> bnfin -> gemm(64->128)
//   -> bnfin -> voxel hist -> stable counting-sort scan -> scatter(sel) ->
//   gather centers -> ball candidate compaction -> rank-select 32-NN
//   + channel max-pool.
// All WMMA operands are contiguous aligned vector loads (no divergence);
// activations stored [N, C]; BN stats fused into the GEMMs via LDS atomics.
// ---------------------------------------------------------------------------

#define NPTS    65536
#define SCENT   2048
#define MAXCAND 1024
#define R2F     0.04f
#define STEPV   (2.0f / 8.0f + 0.0002f)
#define OFFV    1.0011f

typedef __attribute__((ext_vector_type(16))) _Float16 v16h;
typedef __attribute__((ext_vector_type(8)))  float    v8f;

// ---------------- init: zero BN accumulators (sum+sumsq, 32+64+128 ch) -----
__global__ void init_ws(float* __restrict__ sums) {
    int i = blockIdx.x * blockDim.x + threadIdx.x;
    if (i < 448) sums[i] = 0.0f;
}

// ---------------- pack weights into WMMA A-register order f16 tiles --------
// dst layout: [tile t = mt*NK + kt][lane 0..31][h 0..15], zero-padded K>=Cin.
__global__ __launch_bounds__(512) void pack_weights(
    const float* __restrict__ W, _Float16* __restrict__ dst,
    int Cin, int CinPad) {
    int t    = blockIdx.x;
    int NK   = CinPad >> 5;
    int mt   = t / NK;
    int kt   = t - mt * NK;
    int lane = threadIdx.x >> 4;
    int h    = threadIdx.x & 15;
    int half = lane >> 4;
    int mA   = lane & 15;
    int kb   = half * 8;
    int v = h >> 1, w = h & 1;
    int k = kt * 32 + 2 * kb + 2 * v + w;
    int m = mt * 16 + mA;
    float f = (k < Cin) ? W[m * Cin + k] : 0.0f;
    dst[((size_t)t * 32 + lane) * 16 + h] = (_Float16)f;
}

// ---------------- pack feats = [xyz;points] as f16 [N,32], zero-padded -----
__global__ __launch_bounds__(256) void pack_feats(
    const float* __restrict__ xyz, const float* __restrict__ points,
    _Float16* __restrict__ dst) {
    int i = blockIdx.x * 256 + threadIdx.x;
    _Float16* row = dst + (size_t)i * 32;
    row[0] = (_Float16)xyz[i];
    row[1] = (_Float16)xyz[NPTS + i];
    row[2] = (_Float16)xyz[2 * NPTS + i];
    row[3] = (_Float16)points[i];
    row[4] = (_Float16)points[NPTS + i];
    row[5] = (_Float16)points[2 * NPTS + i];
#pragma unroll
    for (int k = 6; k < 32; ++k) row[k] = (_Float16)0.0f;
}

// ---------------- layer 0: packed f16 B, packed A tiles, WMMA --------------
__global__ __launch_bounds__(256) void gemm0_wmma(
    const _Float16* __restrict__ featsH /*[N,32] f16*/,
    const _Float16* __restrict__ wp     /*2 packed A tiles*/,
    const float* __restrict__ bias,
    float* __restrict__ Yout /*[N,32] pre-BN*/, float* __restrict__ sums) {
    __shared__ float s_sum[32], s_sq[32];
    if (threadIdx.x < 32) { s_sum[threadIdx.x] = 0.f; s_sq[threadIdx.x] = 0.f; }
    __syncthreads();

    const int lane = threadIdx.x & 31;
    const int wave = threadIdx.x >> 5;
    const int half = lane >> 4;
    const int n    = blockIdx.x * 128 + wave * 16 + (lane & 15);

    const v16h b = *(const v16h*)(featsH + (size_t)n * 32 + half * 16);
#pragma unroll
    for (int mt = 0; mt < 2; ++mt) {
        const v16h a = *(const v16h*)(wp + ((size_t)mt * 32 + lane) * 16);
        const int mrow = mt * 16 + half * 8;
        v8f c;
#pragma unroll
        for (int r = 0; r < 8; ++r) c[r] = bias[mrow + r];
        c = __builtin_amdgcn_wmma_f32_16x16x32_f16(false, a, false, b,
                                                   (short)0, c, false, false);
        float* dst = Yout + (size_t)n * 32 + mrow;
#pragma unroll
        for (int r = 0; r < 8; ++r) {
            float vv = c[r];
            dst[r] = vv;
            atomicAdd(&s_sum[mrow + r], vv);
            atomicAdd(&s_sq[mrow + r], vv * vv);
        }
    }
    __syncthreads();
    if (threadIdx.x < 32) {
        atomicAdd(&sums[threadIdx.x],      s_sum[threadIdx.x]);
        atomicAdd(&sums[32 + threadIdx.x], s_sq[threadIdx.x]);
    }
}

// ---------------- mid layers: BN+ReLU fused into B-load, packed A ----------
template <int CIN, int COUT>
__global__ __launch_bounds__(256) void gemm_mid_wmma(
    const float* __restrict__ Xin /*[N,CIN] pre-BN*/,
    const _Float16* __restrict__ wp /*packed A tiles [mt*NK+kt]*/,
    const float* __restrict__ bias,
    const float* __restrict__ mean, const float* __restrict__ rstd,
    const float* __restrict__ g,    const float* __restrict__ bt,
    float* __restrict__ Yout /*[N,COUT] pre-BN*/, float* __restrict__ sums) {
    __shared__ float s_sum[COUT], s_sq[COUT];
    for (int i = threadIdx.x; i < COUT; i += 256) { s_sum[i] = 0.f; s_sq[i] = 0.f; }
    __syncthreads();

    const int lane = threadIdx.x & 31;
    const int wave = threadIdx.x >> 5;
    const int half = lane >> 4;
    const int n    = blockIdx.x * 128 + wave * 16 + (lane & 15);

    constexpr int NK = CIN / 32;
    v16h btile[NK];                             // B per lane: 16 contiguous K
#pragma unroll
    for (int kt = 0; kt < NK; ++kt) {
        const int kbase = kt * 32 + half * 16;
        const float* row = Xin + (size_t)n * CIN + kbase;
        v16h bv;
#pragma unroll
        for (int e = 0; e < 16; ++e) {
            int k = kbase + e;
            float f = row[e];
            f = (f - mean[k]) * rstd[k] * g[k] + bt[k];
            bv[e] = (_Float16)fmaxf(f, 0.0f);
        }
        btile[kt] = bv;
    }

#pragma unroll
    for (int mt = 0; mt < COUT / 16; ++mt) {
        const int mrow = mt * 16 + half * 8;
        v8f c;
#pragma unroll
        for (int r = 0; r < 8; ++r) c[r] = bias[mrow + r];
#pragma unroll
        for (int kt = 0; kt < NK; ++kt) {
            const v16h a =
                *(const v16h*)(wp + ((size_t)(mt * NK + kt) * 32 + lane) * 16);
            c = __builtin_amdgcn_wmma_f32_16x16x32_f16(false, a, false, btile[kt],
                                                       (short)0, c, false, false);
        }
        float* dst = Yout + (size_t)n * COUT + mrow;
#pragma unroll
        for (int r = 0; r < 8; ++r) {
            float vv = c[r];
            dst[r] = vv;
            atomicAdd(&s_sum[mrow + r], vv);
            atomicAdd(&s_sq[mrow + r], vv * vv);
        }
    }
    __syncthreads();
    for (int i = threadIdx.x; i < COUT; i += 256) {
        atomicAdd(&sums[i],        s_sum[i]);
        atomicAdd(&sums[COUT + i], s_sq[i]);
    }
}

// ---------------- BN finalize: mean / rsqrt(var+eps) -----------------------
__global__ void bn_finalize(const float* __restrict__ sums,
                            float* __restrict__ mean, float* __restrict__ rstd,
                            int C, float invN) {
    int c = blockIdx.x * blockDim.x + threadIdx.x;
    if (c < C) {
        float m = sums[c] * invN;
        float v = sums[C + c] * invN - m * m;
        mean[c] = m;
        rstd[c] = rsqrtf(v + 1e-5f);
    }
}

// ---------------- voxel keys + per-block histograms ------------------------
__global__ __launch_bounds__(256) void voxel_hist(
    const float* __restrict__ xyz, int* __restrict__ vkeys, int* __restrict__ hist) {
    __shared__ int h[512];
    for (int i = threadIdx.x; i < 512; i += 256) h[i] = 0;
    __syncthreads();
    int i = blockIdx.x * 256 + threadIdx.x;
    float px = xyz[i], py = xyz[NPTS + i], pz = xyz[2 * NPTS + i];
    int vx = min(max((int)floorf((px + OFFV) / STEPV), 0), 7);
    int vy = min(max((int)floorf((py + OFFV) / STEPV), 0), 7);
    int vz = min(max((int)floorf((pz + OFFV) / STEPV), 0), 7);
    int key = (vx * 8 + vy) * 8 + vz;
    vkeys[i] = key;
    atomicAdd(&h[key], 1);
    __syncthreads();
    for (int k = threadIdx.x; k < 512; k += 256) hist[blockIdx.x * 512 + k] = h[k];
}

// ---------------- scan: hist[b][k] -> stable global base per (block,key) ---
__global__ void hist_scan(int* __restrict__ hist, int nblocks) {
    __shared__ int keyTotal[512];
    __shared__ int keyStart[512];
    int k = threadIdx.x;                        // 512 threads, one key each
    int run = 0;
    for (int b = 0; b < nblocks; ++b) {
        int h = hist[b * 512 + k];
        hist[b * 512 + k] = run;                // exclusive prefix within key
        run += h;
    }
    keyTotal[k] = run;
    __syncthreads();
    if (threadIdx.x == 0) {
        int acc = 0;
        for (int i = 0; i < 512; ++i) { keyStart[i] = acc; acc += keyTotal[i]; }
    }
    __syncthreads();
    int base = keyStart[k];
    for (int b = 0; b < nblocks; ++b) hist[b * 512 + k] += base;
}

// ---------------- stable scatter: emit order[s*32] as sel[s] ---------------
__global__ __launch_bounds__(256) void scatter_sel(
    const int* __restrict__ vkeys, const int* __restrict__ base, int* __restrict__ sel) {
    __shared__ int skeys[256];
    int i = blockIdx.x * 256 + threadIdx.x;
    int k = vkeys[i];
    skeys[threadIdx.x] = k;
    __syncthreads();
    int lr = 0;                                  // stable intra-block rank
    for (int j = 0; j < threadIdx.x; ++j) lr += (skeys[j] == k);
    int pos = base[blockIdx.x * 512 + k] + lr;   // global stable sorted position
    if ((pos & 31) == 0) sel[pos >> 5] = i;
}

// ---------------- gather centers -> out[0:3*2048] --------------------------
__global__ void gather_centers(const float* __restrict__ xyz,
                               const int* __restrict__ sel, float* __restrict__ out) {
    int s = blockIdx.x * blockDim.x + threadIdx.x;
    if (s < SCENT) {
        int i = sel[s];
        out[s]             = xyz[i];
        out[SCENT + s]     = xyz[NPTS + i];
        out[2 * SCENT + s] = xyz[2 * NPTS + i];
    }
}

// ---------------- in-radius candidate compaction ---------------------------
__global__ __launch_bounds__(256) void ball_select(
    const float* __restrict__ xyz, const float* __restrict__ centers /*out[0:6144]*/,
    float* __restrict__ cd2, int* __restrict__ cidx, int* __restrict__ ccnt) {
    int s = blockIdx.x;
    float cx = centers[s], cy = centers[SCENT + s], cz = centers[2 * SCENT + s];
    __shared__ int cnt;
    if (threadIdx.x == 0) cnt = 0;
    __syncthreads();
    const float* X = xyz;
    const float* Y = xyz + NPTS;
    const float* Z = xyz + 2 * NPTS;
    for (int i = threadIdx.x; i < NPTS; i += 256) {
        float dx = X[i] - cx, dy = Y[i] - cy, dz = Z[i] - cz;
        float d2 = dx * dx + dy * dy + dz * dz;
        if (d2 <= R2F) {
            int p = atomicAdd(&cnt, 1);
            if (p < MAXCAND) {
                cd2[(size_t)s * MAXCAND + p]  = d2;
                cidx[(size_t)s * MAXCAND + p] = i;
            }
        }
    }
    __syncthreads();
    if (threadIdx.x == 0) ccnt[s] = min(cnt, MAXCAND);
}

// ---------------- rank-select 32 nearest + channel max-pool ----------------
__global__ __launch_bounds__(128) void select_aggregate(
    const float* __restrict__ cd2, const int* __restrict__ cidx,
    const int* __restrict__ ccnt,  const float* __restrict__ y2 /*[N,128]*/,
    const float* __restrict__ mean2, const float* __restrict__ rstd2,
    const float* __restrict__ g2,    const float* __restrict__ bt2,
    float* __restrict__ out_pts /*[128][2048]*/) {
    __shared__ float sd2[MAXCAND];
    __shared__ int   sidx[MAXCAND];
    __shared__ int   ssel[64];
    __shared__ int   nsel;
    int s   = blockIdx.x;
    int cnt = ccnt[s];
    for (int j = threadIdx.x; j < cnt; j += 128) {
        sd2[j]  = cd2[(size_t)s * MAXCAND + j];
        sidx[j] = cidx[(size_t)s * MAXCAND + j];
    }
    if (threadIdx.x == 0) nsel = 0;
    __syncthreads();
    if (cnt <= 32) {
        for (int j = threadIdx.x; j < cnt; j += 128) {
            int p = atomicAdd(&nsel, 1);
            if (p < 64) ssel[p] = sidx[j];
        }
    } else {
        for (int j = threadIdx.x; j < cnt; j += 128) {
            float dj = sd2[j]; int ij = sidx[j]; int rank = 0;
            for (int t = 0; t < cnt; ++t) {
                float dt = sd2[t];
                rank += (dt < dj) || (dt == dj && sidx[t] < ij);
            }
            if (rank < 32) {                    // among 32 nearest (stable ties)
                int p = atomicAdd(&nsel, 1);
                if (p < 64) ssel[p] = ij;
            }
        }
    }
    __syncthreads();
    int m = min(nsel, 64);                      // center itself guarantees m>=1
    int c = threadIdx.x;                        // one channel per thread: coalesced
    float acc = -3.402823466e38f;
    for (int j = 0; j < m; ++j) {
        float v = y2[(size_t)ssel[j] * 128 + c];
        v = (v - mean2[c]) * rstd2[c] * g2[c] + bt2[c];
        v = fmaxf(v, 0.0f);
        acc = fmaxf(acc, v);
    }
    out_pts[c * SCENT + s] = acc;
}

// ---------------------------------------------------------------------------
extern "C" void kernel_launch(void* const* d_in, const int* in_sizes, int n_in,
                              void* d_out, int out_size, void* d_ws, size_t ws_size,
                              hipStream_t stream) {
    const float* xyz    = (const float*)d_in[0];
    const float* points = (const float*)d_in[1];
    const float* w0  = (const float*)d_in[2];
    const float* b0  = (const float*)d_in[3];
    const float* g0  = (const float*)d_in[4];
    const float* bt0 = (const float*)d_in[5];
    const float* w1  = (const float*)d_in[6];
    const float* b1  = (const float*)d_in[7];
    const float* g1  = (const float*)d_in[8];
    const float* bt1 = (const float*)d_in[9];
    const float* w2  = (const float*)d_in[10];
    const float* b2  = (const float*)d_in[11];
    const float* g2  = (const float*)d_in[12];
    const float* bt2 = (const float*)d_in[13];
    float* out = (float*)d_out;

    char* ws = (char*)d_ws;
    size_t off = 0;
    auto alloc = [&](size_t bytes) -> void* {
        void* p = ws + off;
        off = (off + bytes + 255) & ~(size_t)255;
        return p;
    };
    float*     y0     = (float*)alloc((size_t)NPTS * 32 * 4);  // reused as cd2
    float*     y1     = (float*)alloc((size_t)NPTS * 64 * 4);  // reused as cidx
    float*     y2     = (float*)alloc((size_t)NPTS * 128 * 4);
    _Float16*  featsH = (_Float16*)alloc((size_t)NPTS * 32 * 2);
    _Float16*  wp0    = (_Float16*)alloc((size_t)2 * 512 * 2);   // 2 tiles
    _Float16*  wp1    = (_Float16*)alloc((size_t)4 * 512 * 2);   // 4 tiles
    _Float16*  wp2    = (_Float16*)alloc((size_t)16 * 512 * 2);  // 16 tiles
    float*     sums   = (float*)alloc(448 * 4);
    float*     stats  = (float*)alloc(448 * 4);
    int*       vkeys  = (int*)alloc((size_t)NPTS * 4);
    int*       hist   = (int*)alloc((size_t)256 * 512 * 4);
    int*       sel    = (int*)alloc((size_t)SCENT * 4);
    int*       ccnt   = (int*)alloc((size_t)SCENT * 4);
    float*     cd2    = y0;                                    // overlay dead y0
    int*       cidx   = (int*)y1;                              // overlay dead y1

    float* mean0 = stats;        float* rstd0 = stats + 32;
    float* mean1 = stats + 64;   float* rstd1 = stats + 128;
    float* mean2 = stats + 192;  float* rstd2 = stats + 320;
    const float invN = 1.0f / (float)NPTS;

    init_ws<<<2, 256, 0, stream>>>(sums);
    pack_feats<<<NPTS / 256, 256, 0, stream>>>(xyz, points, featsH);
    pack_weights<<<2, 512, 0, stream>>>(w0, wp0, 6, 32);
    pack_weights<<<4, 512, 0, stream>>>(w1, wp1, 32, 32);
    pack_weights<<<16, 512, 0, stream>>>(w2, wp2, 64, 64);

    gemm0_wmma<<<NPTS / 128, 256, 0, stream>>>(featsH, wp0, b0, y0, sums);
    bn_finalize<<<1, 128, 0, stream>>>(sums, mean0, rstd0, 32, invN);

    gemm_mid_wmma<32, 64><<<NPTS / 128, 256, 0, stream>>>(
        y0, wp1, b1, mean0, rstd0, g0, bt0, y1, sums + 64);
    bn_finalize<<<1, 128, 0, stream>>>(sums + 64, mean1, rstd1, 64, invN);

    gemm_mid_wmma<64, 128><<<NPTS / 128, 256, 0, stream>>>(
        y1, wp2, b2, mean1, rstd1, g1, bt1, y2, sums + 192);
    bn_finalize<<<1, 128, 0, stream>>>(sums + 192, mean2, rstd2, 128, invN);

    voxel_hist<<<NPTS / 256, 256, 0, stream>>>(xyz, vkeys, hist);
    hist_scan<<<1, 512, 0, stream>>>(hist, NPTS / 256);
    scatter_sel<<<NPTS / 256, 256, 0, stream>>>(vkeys, hist, sel);
    gather_centers<<<SCENT / 256, 256, 0, stream>>>(xyz, sel, out);

    ball_select<<<SCENT, 256, 0, stream>>>(xyz, out, cd2, cidx, ccnt);
    select_aggregate<<<SCENT, 128, 0, stream>>>(
        cd2, cidx, ccnt, y2, mean2, rstd2, g2, bt2, out + 3 * SCENT);
}
